// MoELayer_57844619543129
// MI455X (gfx1250) — matrix operand order repaired and checked by reading, not compiled, not run on hardware
//
#include <hip/hip_runtime.h>

typedef __attribute__((ext_vector_type(16))) _Float16      v16h;
typedef __attribute__((ext_vector_type(8)))  float         v8f;
typedef __attribute__((ext_vector_type(4)))  unsigned int  u32x4;
typedef __attribute__((ext_vector_type(2)))  unsigned int  u32x2;
typedef __attribute__((ext_vector_type(4)))  float         f32x4;
typedef __attribute__((__vector_size__(4 * sizeof(int)))) int vi4;  // matches builtin param

#define T_TOK 2048
#define DDIM  1024
#define NEXP  8
#define FMOE  512
#define FSH   2048
#define HCOLS (NEXP * FMOE + FSH)   // 6144 concatenated intermediate columns
#define KB    32                    // K depth per WMMA f16 step

// ---- async global->LDS copy (gfx1250 ASYNCcnt path), with fallback --------
#if __has_builtin(__builtin_amdgcn_global_load_async_to_lds_b128) && \
    __has_builtin(__builtin_amdgcn_s_wait_asynccnt)
#define USE_ASYNC_LDS 1
#define ASYNC_CP16(gsrc, ldst)                                              \
  __builtin_amdgcn_global_load_async_to_lds_b128(                           \
      (vi4*)(gsrc), (vi4*)(ldst), 0, 0)
#define ASYNC_WAIT() __builtin_amdgcn_s_wait_asynccnt(0)
#else
#define USE_ASYNC_LDS 0
#define ASYNC_WAIT()
#endif

// ---------------------------------------------------------------------------
// 1) Router: fp32 logits -> softmax -> top-2 -> renorm -> dense combine [T,8]
// ---------------------------------------------------------------------------
__global__ __launch_bounds__(256) void router_kernel(
    const float* __restrict__ x, const float* __restrict__ rw,
    float* __restrict__ combine)
{
  int t = blockIdx.x * blockDim.x + threadIdx.x;
  if (t >= T_TOK) return;

  float lg[NEXP];
#pragma unroll
  for (int e = 0; e < NEXP; ++e) lg[e] = 0.f;

  const float* xr = x + (size_t)t * DDIM;
  for (int d = 0; d < DDIM; ++d) {
    float xv = xr[d];
    const float* w = rw + (size_t)d * NEXP;
#pragma unroll
    for (int e = 0; e < NEXP; ++e) lg[e] = __builtin_fmaf(xv, w[e], lg[e]);
  }

  float mx = lg[0];
#pragma unroll
  for (int e = 1; e < NEXP; ++e) mx = fmaxf(mx, lg[e]);
  float p[NEXP], s = 0.f;
#pragma unroll
  for (int e = 0; e < NEXP; ++e) { p[e] = __expf(lg[e] - mx); s += p[e]; }
  float inv = 1.f / s;
#pragma unroll
  for (int e = 0; e < NEXP; ++e) p[e] *= inv;

  // top-2 (ties -> lowest index, matching lax.top_k)
  int i1 = 0;
#pragma unroll
  for (int e = 1; e < NEXP; ++e) if (p[e] > p[i1]) i1 = e;
  int i2 = (i1 == 0) ? 1 : 0;
#pragma unroll
  for (int e = 0; e < NEXP; ++e) if (e != i1 && p[e] > p[i2]) i2 = e;

  float w1 = p[i1], w2 = p[i2];
  float wn = 1.f / (w1 + w2);   // NORM_TOPK
  w1 *= wn; w2 *= wn;

  float* c = combine + (size_t)t * NEXP;
#pragma unroll
  for (int e = 0; e < NEXP; ++e)
    c[e] = (e == i1) ? w1 : ((e == i2) ? w2 : 0.f);
}

// ---------------------------------------------------------------------------
// 2) Cast x fp32 -> f16 (A matrix shared by all GEMMs)
// ---------------------------------------------------------------------------
__global__ __launch_bounds__(256) void cast_kernel(
    const float* __restrict__ x, _Float16* __restrict__ xh, int n4)
{
  int i = blockIdx.x * blockDim.x + threadIdx.x;
  if (i >= n4) return;
  f32x4 v = ((const f32x4*)x)[i];
  union { _Float16 h[4]; u32x2 q; } o;
#pragma unroll
  for (int k = 0; k < 4; ++k) o.h[k] = (_Float16)v[k];
  *(u32x2*)(xh + (size_t)i * 4) = o.q;
}

// ---------------------------------------------------------------------------
// 3) Gate/Up GEMM + SwiGLU + combine-scale, for 8 experts + shared unit.
//    Block = 128 threads (4 waves), tile 64x64, K-step 32, f16 WMMA.
// ---------------------------------------------------------------------------
__global__ __launch_bounds__(128) void gateup_kernel(
    const _Float16* __restrict__ xh,
    const float* __restrict__ eg, const float* __restrict__ eu,
    const float* __restrict__ sg, const float* __restrict__ su,
    const float* __restrict__ combine,
    _Float16* __restrict__ hcat)
{
  const int unit = blockIdx.z;                 // 0..7 experts, 8 = shared
  const bool is_sh = (unit == NEXP);
  const int F   = is_sh ? FSH : FMOE;
  const int tn  = blockIdx.y;
  if (tn * 64 >= F) return;                    // uniform per block
  const int tm  = blockIdx.x;
  const float* gptr = is_sh ? sg : eg + (size_t)unit * DDIM * FMOE;
  const float* uptr = is_sh ? su : eu + (size_t)unit * DDIM * FMOE;
  const int ldb   = F;
  const int hbase = is_sh ? NEXP * FMOE : unit * FMOE;

  __shared__ _Float16 As[64 * KB];             // row-major [m][k]
  __shared__ _Float16 Bg[64 * KB];             // transposed [n][k]
  __shared__ _Float16 Bu[64 * KB];

  const int tid  = threadIdx.x;
  const int lane = tid & 31;
  const int wave = tid >> 5;

  v8f accg[4] = {};
  v8f accu[4] = {};

  // A-tile load mapping: 2 x 16B per thread
  const int ar = tid >> 1;
  const int ah = (tid & 1) * 16;               // f16 element offset in row
  const size_t arow = (size_t)(tm * 64 + ar) * DDIM;
  // B-tile load mapping: thread owns (col bn, 16 k-rows starting at bk)
  const int bn = tid & 63;
  const int bk = (tid >> 6) * 16;

  // fragment addressing (contiguous k = 16*(lane/16) + s, same order in A and B)
  const int mrow = wave * 16 + (lane & 15);
  const int akoff = (lane < 16) ? 0 : 16;
  const int bkoff = (lane < 16) ? 0 : 16;

  for (int ks = 0; ks < DDIM / KB; ++ks) {
    // ---- stage A (already f16): async copy straight into LDS ----
#if USE_ASYNC_LDS
    ASYNC_CP16(xh + arow + ks * KB + ah,     As + ar * KB + ah);
    ASYNC_CP16(xh + arow + ks * KB + ah + 8, As + ar * KB + ah + 8);
#else
    {
      const u32x4* src = (const u32x4*)(xh + arow + ks * KB + ah);
      u32x4 a0 = src[0], a1 = src[1];
      *(u32x4*)(As + ar * KB + ah)     = a0;
      *(u32x4*)(As + ar * KB + ah + 8) = a1;
    }
#endif
    // ---- stage B gate/up: fp32 strided loads -> f16 transposed tile ----
    {
      const float* gc = gptr + (size_t)(ks * KB + bk) * ldb + tn * 64 + bn;
      const float* uc = uptr + (size_t)(ks * KB + bk) * ldb + tn * 64 + bn;
      if (ks + 1 < DDIM / KB) {                // prefetch next K tile
        __builtin_prefetch(gc + (size_t)KB * ldb, 0, 1);
        __builtin_prefetch(uc + (size_t)KB * ldb, 0, 1);
      }
      union { _Float16 h[16]; u32x4 q[2]; } tg, tu;
#pragma unroll
      for (int i = 0; i < 16; ++i) {
        tg.h[i] = (_Float16)gc[(size_t)i * ldb];
        tu.h[i] = (_Float16)uc[(size_t)i * ldb];
      }
      *(u32x4*)(Bg + bn * KB + bk)     = tg.q[0];
      *(u32x4*)(Bg + bn * KB + bk + 8) = tg.q[1];
      *(u32x4*)(Bu + bn * KB + bk)     = tu.q[0];
      *(u32x4*)(Bu + bn * KB + bk + 8) = tu.q[1];
    }
    ASYNC_WAIT();
    __syncthreads();

    union { v16h v; u32x4 q[2]; } af;
    af.q[0] = *(const u32x4*)(As + mrow * KB + akoff);
    af.q[1] = *(const u32x4*)(As + mrow * KB + akoff + 8);
#pragma unroll
    for (int j = 0; j < 4; ++j) {
      const int ncol = j * 16 + (lane & 15);
      union { v16h v; u32x4 q[2]; } bf;
      bf.q[0] = *(const u32x4*)(Bg + ncol * KB + bkoff);
      bf.q[1] = *(const u32x4*)(Bg + ncol * KB + bkoff + 8);
      accg[j] = __builtin_amdgcn_wmma_f32_16x16x32_f16(
          false, af.v, false, bf.v, (short)0, accg[j], false, false);
      bf.q[0] = *(const u32x4*)(Bu + ncol * KB + bkoff);
      bf.q[1] = *(const u32x4*)(Bu + ncol * KB + bkoff + 8);
      accu[j] = __builtin_amdgcn_wmma_f32_16x16x32_f16(
          false, af.v, false, bf.v, (short)0, accu[j], false, false);
    }
    __syncthreads();
  }

  // ---- epilogue: silu(g)*u * combine  -> f16 Hcat ----
  const int rbase = tm * 64 + wave * 16 + ((lane < 16) ? 0 : 8);
  float sc[8];
#pragma unroll
  for (int r = 0; r < 8; ++r)
    sc[r] = is_sh ? 1.0f : combine[(size_t)(rbase + r) * NEXP + unit];

#pragma unroll
  for (int j = 0; j < 4; ++j) {
    const int col = hbase + tn * 64 + j * 16 + (lane & 15);
#pragma unroll
    for (int r = 0; r < 8; ++r) {
      float g = accg[j][r], u = accu[j][r];
      float h = (g / (1.f + __expf(-g))) * u * sc[r];
      hcat[(size_t)(rbase + r) * HCOLS + col] = (_Float16)h;
    }
  }
}

// ---------------------------------------------------------------------------
// 4) Down projection: out = Hcat @ concat(experts_down, shared_down), K=6144
// ---------------------------------------------------------------------------
__global__ __launch_bounds__(128) void down_kernel(
    const _Float16* __restrict__ hcat,
    const float* __restrict__ ed, const float* __restrict__ sd,
    float* __restrict__ out)
{
  const int tm = blockIdx.x, tn = blockIdx.y;

  __shared__ _Float16 As[64 * KB];
  __shared__ _Float16 Bt[64 * KB];

  const int tid  = threadIdx.x;
  const int lane = tid & 31;
  const int wave = tid >> 5;

  v8f acc[4] = {};

  const int ar = tid >> 1;
  const int ah = (tid & 1) * 16;
  const size_t arow = (size_t)(tm * 64 + ar) * HCOLS;
  const int bn = tid & 63;
  const int bk = (tid >> 6) * 16;

  const int mrow  = wave * 16 + (lane & 15);
  const int akoff = (lane < 16) ? 0 : 16;
  const int bkoff = (lane < 16) ? 0 : 16;

  const int KSTEPS_E = (NEXP * FMOE) / KB;     // 128
  for (int ks = 0; ks < HCOLS / KB; ++ks) {    // 192
#if USE_ASYNC_LDS
    ASYNC_CP16(hcat + arow + ks * KB + ah,     As + ar * KB + ah);
    ASYNC_CP16(hcat + arow + ks * KB + ah + 8, As + ar * KB + ah + 8);
#else
    {
      const u32x4* src = (const u32x4*)(hcat + arow + ks * KB + ah);
      u32x4 a0 = src[0], a1 = src[1];
      *(u32x4*)(As + ar * KB + ah)     = a0;
      *(u32x4*)(As + ar * KB + ah + 8) = a1;
    }
#endif
    {
      // select weight source for this K block of the concatenated matrix
      const float* bsrc;
      if (ks < KSTEPS_E) {
        int e = ks >> 4;                              // 16 K-blocks per expert
        bsrc = ed + (size_t)e * FMOE * DDIM + (size_t)((ks & 15) * KB) * DDIM;
      } else {
        bsrc = sd + (size_t)((ks - KSTEPS_E) * KB) * DDIM;
      }
      const float* c = bsrc + (size_t)bk * DDIM + tn * 64 + bn;
      __builtin_prefetch(c + (size_t)KB * DDIM, 0, 1);
      union { _Float16 h[16]; u32x4 q[2]; } tb;
#pragma unroll
      for (int i = 0; i < 16; ++i) tb.h[i] = (_Float16)c[(size_t)i * DDIM];
      *(u32x4*)(Bt + bn * KB + bk)     = tb.q[0];
      *(u32x4*)(Bt + bn * KB + bk + 8) = tb.q[1];
    }
    ASYNC_WAIT();
    __syncthreads();

    union { v16h v; u32x4 q[2]; } af;
    af.q[0] = *(const u32x4*)(As + mrow * KB + akoff);
    af.q[1] = *(const u32x4*)(As + mrow * KB + akoff + 8);
#pragma unroll
    for (int j = 0; j < 4; ++j) {
      const int ncol = j * 16 + (lane & 15);
      union { v16h v; u32x4 q[2]; } bf;
      bf.q[0] = *(const u32x4*)(Bt + ncol * KB + bkoff);
      bf.q[1] = *(const u32x4*)(Bt + ncol * KB + bkoff + 8);
      acc[j] = __builtin_amdgcn_wmma_f32_16x16x32_f16(
          false, af.v, false, bf.v, (short)0, acc[j], false, false);
    }
    __syncthreads();
  }

  const int rbase = tm * 64 + wave * 16 + ((lane < 16) ? 0 : 8);
#pragma unroll
  for (int j = 0; j < 4; ++j) {
    const int col = tn * 64 + j * 16 + (lane & 15);
#pragma unroll
    for (int r = 0; r < 8; ++r)
      out[(size_t)(rbase + r) * DDIM + col] = acc[j][r];
  }
}

// ---------------------------------------------------------------------------
extern "C" void kernel_launch(void* const* d_in, const int* in_sizes, int n_in,
                              void* d_out, int out_size, void* d_ws, size_t ws_size,
                              hipStream_t stream)
{
  const float* x  = (const float*)d_in[0];   // [T, D]
  const float* rw = (const float*)d_in[1];   // [D, E]
  const float* eg = (const float*)d_in[2];   // [E, D, Fm]
  const float* eu = (const float*)d_in[3];   // [E, D, Fm]
  const float* ed = (const float*)d_in[4];   // [E, Fm, D]
  const float* sg = (const float*)d_in[5];   // [D, Fs]
  const float* su = (const float*)d_in[6];   // [D, Fs]
  const float* sd = (const float*)d_in[7];   // [Fs, D]
  float* out = (float*)d_out;                // [T, D] fp32

  char* ws = (char*)d_ws;
  float*    combine = (float*)ws;                                   // 64 KB
  _Float16* xh      = (_Float16*)(ws + 65536);                      // 4 MB
  _Float16* hcat    = (_Float16*)(ws + 65536 + (size_t)T_TOK * DDIM * 2); // 24 MB

  router_kernel<<<T_TOK / 256, 256, 0, stream>>>(x, rw, combine);
  cast_kernel<<<(T_TOK * DDIM / 4) / 256, 256, 0, stream>>>(x, xh, T_TOK * DDIM / 4);

  dim3 g1(T_TOK / 64, FSH / 64, NEXP + 1);   // y over max(F)/64; experts early-exit
  gateup_kernel<<<g1, 128, 0, stream>>>(xh, eg, eu, sg, su, combine, hcat);

  dim3 g2(T_TOK / 64, DDIM / 64);
  down_kernel<<<g2, 128, 0, stream>>>(hcat, ed, sd, out);
}
